// ConvLayer_26371099197994
// MI455X (gfx1250) — matrix operand order repaired and checked by reading, not compiled, not run on hardware
//
#include <hip/hip_runtime.h>
#include <hip/hip_bf16.h>

// MI455X / gfx1250 fused StyleGAN2 up-conv layer.
// - Transposed conv (up=2, 3x3) + FIR [1,3,3,1]^2 folded into 4 parity 3x3 convs
//   (exact Winograd-style weight transform) -> v_wmma_f32_16x16x32_f16.
// - Tensor Data Mover (TENSOR_LOAD_TO_LDS + s_wait_tensorcnt) streams x tiles and
//   weight tiles into double-buffered LDS, overlapping DMA with WMMA.
// d_ws layout: [0, 18.9MB) transformed weights f16 | [18.9MB, 54.6MB) padded NHWC x f16.

typedef __attribute__((ext_vector_type(16))) _Float16 v16h;
typedef __attribute__((ext_vector_type(8)))  float    v8f;
typedef __attribute__((ext_vector_type(4)))  unsigned int v4u;
typedef __attribute__((ext_vector_type(4)))  unsigned int u32x4;
typedef __attribute__((ext_vector_type(8)))  int i32x8;
typedef __attribute__((ext_vector_type(4)))  int i32x4;

union Pack32 { v4u u[2]; v16h h; };

#define CIN   512
#define COUT  512
#define RES   128
#define INRES 64
#define PADR  66          // 64 + 1-pixel zero border each side

// ---------------------------------------------------------------------------
// TDM: issue a 3D-tile load (X=tile0 contiguous, Y stride0, Z stride1; element
// size 2 bytes) from global into LDS, packed densely as [Z][Y][X].
// clang-23 toolchain: 6-arg builtin (g0 x4, g1 x8, g2 x4, g3 x4, x8, cpol).
// ---------------------------------------------------------------------------
__device__ __forceinline__ void tdm_load_3d(
    unsigned lds_addr, const void* gaddr,
    unsigned tile0, unsigned tile1, unsigned tile2,
    unsigned long long stride0, unsigned long long stride1,
    unsigned td0, unsigned td1, unsigned td2)
{
    unsigned long long ga = (unsigned long long)(uintptr_t)gaddr;
    u32x4 g0;
    g0[0] = 1u;                                        // count=1, user mode
    g0[1] = lds_addr;                                  // LDS byte address
    g0[2] = (unsigned)(ga & 0xFFFFFFFFu);              // global addr [31:0]
    g0[3] = (unsigned)((ga >> 32) & 0x01FFFFFFu) | 0x80000000u;  // [56:32] | type=2
    i32x8 g1;
    g1[0] = (int)(1u << 16);                           // data_size=1 (2 bytes)
    g1[1] = (int)((td0 & 0xFFFFu) << 16);              // tensor_dim0 lo
    g1[2] = (int)((td0 >> 16) | ((td1 & 0xFFFFu) << 16));
    g1[3] = (int)((td1 >> 16) | (tile0 << 16));        // tile_dim0
    g1[4] = (int)(tile1 | (tile2 << 16));              // tile_dim1, tile_dim2
    g1[5] = (int)(unsigned)(stride0 & 0xFFFFFFFFull);  // tensor_dim0_stride lo
    g1[6] = (int)((unsigned)((stride0 >> 32) & 0xFFFFull)
                  | ((unsigned)(stride1 & 0xFFFFull) << 16));
    g1[7] = (int)(unsigned)((stride1 >> 16) & 0xFFFFFFFFull);
    i32x4 g2;
    g2[0] = (int)td2;                                  // tensor_dim2
    g2[1] = 0;                                         // tensor_dim3 (unused)
    g2[2] = 0;                                         // tensor_dim2_stride lo
    g2[3] = 0;                                         // stride hi | tile_dim3=0
    i32x4 g3 = {0, 0, 0, 0};
    i32x8 g4 = {0, 0, 0, 0, 0, 0, 0, 0};
    __builtin_amdgcn_tensor_load_to_lds(g0, g1, g2, g3, g4, 0);
}

// ---------------------------------------------------------------------------
// Kernel 1: weight transform  Ct[p][jy][jx][co][ci] = (F_py . flip(w) . F_px^T)
// with style scale 1/sqrt(512*9) folded in. f_eff = [.25,.75,.75,.25].
// ---------------------------------------------------------------------------
__global__ __launch_bounds__(256) void weight_transform(
    const float* __restrict__ w, _Float16* __restrict__ ct)
{
    int idx = blockIdx.x * 256 + threadIdx.x;
    if (idx >= COUT * CIN) return;
    int ci = idx & (CIN - 1);
    int co = idx >> 9;

    const float s = 0.014731391274719739f;            // 1/sqrt(512*9)
    float wf[3][3];
#pragma unroll
    for (int ky = 0; ky < 3; ++ky)
#pragma unroll
        for (int kx = 0; kx < 3; ++kx)
            wf[ky][kx] = w[(size_t)idx * 9 + ky * 3 + kx] * s;

    const float Fe[3][3] = {{0.75f,0.25f,0.f},{0.25f,0.75f,0.75f},{0.f,0.f,0.25f}};
    const float Fo[3][3] = {{0.25f,0.f,0.f},{0.75f,0.75f,0.25f},{0.f,0.25f,0.75f}};

#pragma unroll
    for (int p = 0; p < 4; ++p) {
        const float (*Fy)[3] = (p >> 1) ? Fo : Fe;
        const float (*Fx)[3] = (p & 1) ? Fo : Fe;
#pragma unroll
        for (int jy = 0; jy < 3; ++jy)
#pragma unroll
            for (int jx = 0; jx < 3; ++jx) {
                float a = 0.f;
#pragma unroll
                for (int ky = 0; ky < 3; ++ky)
#pragma unroll
                    for (int kx = 0; kx < 3; ++kx)
                        a += Fy[jy][ky] * Fx[jx][kx] * wf[2 - ky][2 - kx];
                ct[(size_t)((p * 9 + jy * 3 + jx) * COUT + co) * CIN + ci] =
                    (_Float16)a;
            }
    }
}

// ---------------------------------------------------------------------------
// Kernel 2: pack x (f32 NCHW) -> f16 NHWC with 1-pixel zero border:
// xT[n][66][66][512]. LDS transpose keeps reads and writes coalesced.
// Block = (row gyp, 32-ci block, image). grid(66, 16, 8).
// ---------------------------------------------------------------------------
__global__ __launch_bounds__(256) void pack_x(
    const float* __restrict__ x, _Float16* __restrict__ xT)
{
    __shared__ _Float16 row[32][INRES];               // [ci][gx]
    const int gyp = blockIdx.x;                       // 0..65 padded row
    const int ci0 = blockIdx.y * 32;
    const int n   = blockIdx.z;
    const int tid = threadIdx.x;
    const bool borderRow = (gyp == 0) || (gyp == PADR - 1);

    if (!borderRow) {
        const int gy = gyp - 1;
        for (int idx = tid; idx < 32 * INRES; idx += 256) {
            int gx = idx & 63, ci = idx >> 6;         // gx-contiguous reads
            row[ci][gx] = (_Float16)
                x[(((size_t)n * CIN + ci0 + ci) * INRES + gy) * INRES + gx];
        }
    }
    __syncthreads();
    for (int idx = tid; idx < PADR * 32; idx += 256) {
        int ci = idx & 31, gxp = idx >> 5;            // ci-contiguous writes
        _Float16 v = (_Float16)0.f;
        if (!borderRow && gxp >= 1 && gxp <= INRES) v = row[ci][gxp - 1];
        xT[(((size_t)n * PADR + gyp) * PADR + gxp) * CIN + ci0 + ci] = v;
    }
}

// ---------------------------------------------------------------------------
// Kernel 3: fused conv + noise + bias + lrelu + clamp.
// Block: 16 cout x (32x16) output pixels x 1 image; 8 waves; wave parity p=w/2.
// TDM double-buffers x tiles (18x10x32ci) and weight tiles (36x16x32) in LDS.
// ---------------------------------------------------------------------------
__global__ __launch_bounds__(256) void conv_up_fused(
    const _Float16* __restrict__ xT, const _Float16* __restrict__ ct,
    const float* __restrict__ bias, const float* __restrict__ noise,
    const float* __restrict__ nstrp, float* __restrict__ out)
{
    __shared__ __align__(16) _Float16 xs[2][18 * 10 * 32];   // [gy][gx][ci]
    __shared__ __align__(16) _Float16 wA[2][36 * 16 * 32];   // [p*9+tap][co][ci]

    const int tid  = threadIdx.x;
    const int lane = tid & 31;
    const int wv   = tid >> 5;
    const int p    = wv >> 1;
    const int py   = p >> 1, px = p & 1;
    const int tb   = (wv & 1) * 4;
    const int hi   = lane >> 4;
    const int ln   = lane & 15;

    const int tiley = blockIdx.x >> 3, tilex = blockIdx.x & 7;
    const int oy0 = tiley * 32, ox0 = tilex * 16;
    const int m0  = tiley * 16, mx0 = tilex * 8;
    const int co0 = blockIdx.y * 16;
    const int nimg = blockIdx.z;

    const _Float16* xTn = xT + (size_t)nimg * PADR * PADR * CIN;

    auto issue = [&](int cb, int buf) {
        // x tile: X=32 ci, Y=10 gx (stride 512), Z=18 gy (stride 66*512);
        // tile start at padded (m0, mx0) => covers gy in [m0-1, m0+16] etc.
        tdm_load_3d((unsigned)(uintptr_t)(&xs[buf][0]),
                    xTn + ((size_t)m0 * PADR + mx0) * CIN + cb,
                    32, 10, 18, (unsigned long long)CIN,
                    (unsigned long long)PADR * CIN,
                    CIN, 1u << 20, 1u << 20);
        // weight tile: X=32 ci, Y=16 co (stride 512), Z=36 taps (stride 512*512)
        tdm_load_3d((unsigned)(uintptr_t)(&wA[buf][0]),
                    ct + (size_t)co0 * CIN + cb,
                    32, 16, 36, (unsigned long long)CIN,
                    (unsigned long long)COUT * CIN,
                    CIN, 1u << 20, 64);
    };

    v8f acc[4] = {};

    if (wv == 0) { issue(0, 0); __builtin_amdgcn_s_wait_tensorcnt(0); }
    __syncthreads();

    for (int c = 0; c < CIN / 32; ++c) {
        const int buf = c & 1;
        if (wv == 0 && c + 1 < CIN / 32) issue((c + 1) * 32, buf ^ 1);

#pragma unroll
        for (int jy = 0; jy < 3; ++jy)
#pragma unroll
        for (int jx = 0; jx < 3; ++jx) {
            const int pt = p * 9 + jy * 3 + jx;
            Pack32 pa;            // A: lo lanes K[0..8)&[16..24), hi +8
            const _Float16* ab = &wA[buf][(pt * 16 + ln) * 32 + (hi ? 8 : 0)];
            pa.u[0] = *(const v4u*)(ab);
            pa.u[1] = *(const v4u*)(ab + 16);
#pragma unroll
            for (int k = 0; k < 4; ++k) {
                const int t = tb + k;
                const int r = 2 * t + (ln >> 3);
                const int s = ln & 7;
                const _Float16* bb =      // B: lane=column, K[0..8)&[16..24), hi +8
                    &xs[buf][((r + jy) * 10 + (s + jx)) * 32 + hi * 8];
                Pack32 pb;
                pb.u[0] = *(const v4u*)(bb);
                pb.u[1] = *(const v4u*)(bb + 16);
                acc[k] = __builtin_amdgcn_wmma_f32_16x16x32_f16(
                    false, pa.h, false, pb.h, (short)0, acc[k], false, false);
            }
        }

        if (wv == 0) __builtin_amdgcn_s_wait_tensorcnt(0);
        __syncthreads();
    }

    // ---- epilogue: noise + bias + lrelu*sqrt(2) + clamp, store f32 ----
    const float nstr = nstrp[0];
#pragma unroll
    for (int k = 0; k < 4; ++k) {
        const int t = tb + k;
        const int r = 2 * t + (ln >> 3);
        const int s = ln & 7;
        const int oy = oy0 + 2 * r + py;
        const int ox = ox0 + 2 * s + px;
        const float nz = noise[oy * RES + ox] * nstr;
#pragma unroll
        for (int v = 0; v < 8; ++v) {
            const int col = v + hi * 8;               // C layout: M = v + 8*hi
            float val = acc[k][v] + nz + bias[co0 + col];
            val = (val >= 0.f ? val : 0.2f * val) * 1.4142135623730951f;
            val = fminf(fmaxf(val, -256.f), 256.f);
            out[(((size_t)nimg * COUT + co0 + col) * RES + oy) * RES + ox] = val;
        }
    }
}

// ---------------------------------------------------------------------------
extern "C" void kernel_launch(void* const* d_in, const int* in_sizes, int n_in,
                              void* d_out, int out_size, void* d_ws, size_t ws_size,
                              hipStream_t stream) {
    const float* xin   = (const float*)d_in[0];   // [8,512,64,64]
    const float* w     = (const float*)d_in[1];   // [512,512,3,3]
    const float* bias  = (const float*)d_in[2];   // [512]
    const float* noise = (const float*)d_in[3];   // [128,128]
    const float* nstr  = (const float*)d_in[4];   // scalar
    float* out = (float*)d_out;                   // [8,512,128,128] f32

    _Float16* ct = (_Float16*)d_ws;                       // 18.9 MB
    _Float16* xT = ct + (size_t)4 * 9 * COUT * CIN;       // 35.7 MB padded NHWC

    weight_transform<<<(COUT * CIN + 255) / 256, 256, 0, stream>>>(w, ct);
    pack_x<<<dim3(PADR, CIN / 32, 8), 256, 0, stream>>>(xin, xT);

    dim3 grid(32, COUT / 16, 8);
    conv_up_fused<<<grid, 256, 0, stream>>>(xT, ct, bias, noise, nstr, out);
}
